// SoftGroup_31980326486540
// MI455X (gfx1250) — 8-bit matrix operand order repaired and checked by reading, compiled, not run on hardware
//
#include <hip/hip_runtime.h>
#include <hip/hip_bf16.h>

typedef int v8i __attribute__((ext_vector_type(8)));

__device__ __forceinline__ unsigned expand4(unsigned nib) {
    return (nib & 1u) | ((nib & 2u) << 7) | ((nib & 4u) << 14) | ((nib & 8u) << 21);
}

__global__ void sg_init_kernel(unsigned* ws, unsigned long long wsWords,
                               float* out, int outN) {
    unsigned long long i = (unsigned long long)blockIdx.x * blockDim.x + threadIdx.x;
    unsigned long long stride = (unsigned long long)gridDim.x * blockDim.x;
    for (unsigned long long k = i; k < wsWords; k += stride) ws[k] = 0u;
    for (unsigned long long k = i; k < (unsigned long long)outN; k += stride) out[k] = 0.0f;
}

__global__ void sg_refbits_kernel(const int* __restrict__ labels,
                                  const int* __restrict__ objid,
                                  unsigned* __restrict__ refbits,
                                  int totalW, int npoint) {
    int w = blockIdx.x * blockDim.x + threadIdx.x;
    if (w >= totalW) return;
    int n0 = w << 5;
    int b  = n0 / npoint;
    int oid = objid[b];
    unsigned bits = 0u;
    #pragma unroll 4
    for (int t = 0; t < 32; ++t)
        bits |= (labels[n0 + t] == oid) ? (1u << t) : 0u;
    refbits[w] = bits;
}

__global__ void sg_scatter_kernel(const int* __restrict__ pidx,
                                  unsigned* __restrict__ pcbits,
                                  int M, int npoint, int kwords32) {
    int i = blockIdx.x * blockDim.x + threadIdx.x;
    if (i >= M) return;
    int p = pidx[2 * i];
    int j = pidx[2 * i + 1];
    int n = j % npoint;
    atomicOr(&pcbits[(unsigned long long)p * kwords32 + (n >> 5)], 1u << (n & 31));
}

__global__ __launch_bounds__(32)
void sg_iou_wmma_kernel(const unsigned* __restrict__ pcbits,
                        const unsigned* __restrict__ refbits,
                        const int* __restrict__ pes,
                        int* __restrict__ interArr, int* __restrict__ unionArr,
                        int PTOT, int B, int kwords64, int kchunk) {
    const int lane  = threadIdx.x;
    const int half  = lane >> 4;
    const int n     = lane & 15;
    const int tile0 = blockIdx.x * 16;
    const int kb0   = blockIdx.y * kchunk;
    int kb1 = kb0 + kchunk; if (kb1 > kwords64) kb1 = kwords64;

    const int row   = tile0 + n;
    const bool rowOK = row < PTOT;
    const unsigned long long rowC = rowOK ? (unsigned long long)row : 0ull;

    const unsigned long long* pc64 = (const unsigned long long*)pcbits;
    const unsigned long long* rf64 = (const unsigned long long*)refbits;

    const int aoff  = half * 8;
    const int boff  = half * 16;
    const int nb    = (n < B) ? n : (B - 1);
    const bool isRef  = (n < B);
    const bool isOnes = (n == B);

    v8i acc = (v8i){0, 0, 0, 0, 0, 0, 0, 0};

    for (int kb = kb0; kb < kb1; ++kb) {
        unsigned long long w = pc64[rowC * kwords64 + kb];
        w = rowOK ? w : 0ull;
        unsigned long long rw = rf64[(unsigned long long)nb * kwords64 + kb];
        rw = isRef ? rw : (isOnes ? ~0ull : 0ull);

        v8i a, b;
        #pragma unroll
        for (int v = 0; v < 8; ++v) {
            int ka  = ((v >> 1) * 16) + ((v & 1) * 4) + aoff;
            int kbb = ((v >> 1) * 16) + ((v & 1) * 4) + aoff;
            a[v] = (int)expand4((unsigned)((w  >> ka ) & 0xFull));
            b[v] = (int)expand4((unsigned)((rw >> kbb) & 0xFull));
        }
        acc = __builtin_amdgcn_wmma_i32_16x16x64_iu8(false, a, false, b, acc,
                                                     false, false);
    }

    #pragma unroll
    for (int g = 0; g < 8; ++g) {
        int p = tile0 + g + half * 8;
        if (p < PTOT) {
            int bid = 0, accu = 0;
            for (int k = 0; k < B; ++k) { accu += pes[k]; if (p >= accu) bid++; }
            int val = acc[g];
            if (n == bid) atomicAdd(&interArr[p], val);
            if (n == B)   atomicAdd(&unionArr[p], val);
        }
    }
}

__global__ void sg_clus_copy_kernel(const float* __restrict__ feats,
                                    const int* __restrict__ pes,
                                    float* __restrict__ out,
                                    int B, int PTOT, int C, int MAXP) {
    int idx = blockIdx.x * blockDim.x + threadIdx.x;
    if (idx >= PTOT * C) return;
    int p = idx / C, c = idx % C;
    int bid = 0, acc = 0, offb = 0;
    for (int k = 0; k < B; ++k) { acc += pes[k]; if (p >= acc) { bid++; offb = acc; } }
    int slot = p - offb;
    if (slot < MAXP)
        out[((unsigned long long)bid * MAXP + slot) * C + c] = feats[idx];
}

__global__ __launch_bounds__(128)
void sg_finalize_kernel(const unsigned* __restrict__ refbits,
                        const int* __restrict__ interArr,
                        const int* __restrict__ unionArr,
                        const float* __restrict__ feats,
                        const int* __restrict__ pes,
                        float* __restrict__ out,
                        int B, int C, int npoint,
                        int selFeatOff, int selIdxOff, int offOff, int goodOff) {
    __shared__ int   sint[128];
    __shared__ float sval[128];
    __shared__ int   sidx[128];
    __shared__ int   sbest;

    const int b = blockIdx.x;
    const int tid = threadIdx.x;

    int offb = 0;
    for (int k = 0; k < b; ++k) offb += pes[k];
    const int cnt = pes[b];

    const int kwords32 = npoint >> 5;
    int rc = 0;
    for (int w = tid; w < kwords32; w += 128)
        rc += __popc(refbits[b * kwords32 + w]);
    sint[tid] = rc;
    __syncthreads();
    for (int s = 64; s > 0; s >>= 1) {
        if (tid < s) sint[tid] += sint[tid + s];
        __syncthreads();
    }
    const int refc = sint[0];
    __syncthreads();

    float bv = -1.0f; int bi = 0x7FFFFFFF;
    for (int s = tid; s < cnt; s += 128) {
        int p  = offb + s;
        int ic = interArr[p];
        int uc = unionArr[p];
        float u = (float)(uc + refc - ic);
        float iou = (u > 0.0f) ? ((float)ic / fmaxf(u, 1.0f)) : 0.0f;
        if (iou > bv || (iou == bv && p < bi)) { bv = iou; bi = p; }
    }
    sval[tid] = bv; sidx[tid] = bi;
    __syncthreads();
    for (int s = 64; s > 0; s >>= 1) {
        if (tid < s) {
            float v2 = sval[tid + s]; int i2 = sidx[tid + s];
            if (v2 > sval[tid] || (v2 == sval[tid] && i2 < sidx[tid])) {
                sval[tid] = v2; sidx[tid] = i2;
            }
        }
        __syncthreads();
    }

    if (tid == 0) {
        bool has = cnt > 0;
        int best = has ? sidx[0] : -1;
        out[selIdxOff + b] = (float)best;
        out[goodOff + b]   = (has && sval[0] > 0.2f) ? 1.0f : 0.0f;
        sbest = best;
        if (b == 0) {
            int acc = 0;
            out[offOff] = 0.0f;
            for (int k = 0; k < B; ++k) { acc += pes[k]; out[offOff + k + 1] = (float)acc; }
        }
    }
    __syncthreads();
    const int best = sbest;
    for (int c = tid; c < C; c += 128)
        out[selFeatOff + b * C + c] = (best >= 0) ? feats[best * C + c] : 0.0f;
}

extern "C" void kernel_launch(void* const* d_in, const int* in_sizes, int n_in,
                              void* d_out, int out_size, void* d_ws, size_t ws_size,
                              hipStream_t stream) {
    const int*   proposals_idx = (const int*)  d_in[0];
    const int*   pes           = (const int*)  d_in[1];
    const int*   labels        = (const int*)  d_in[2];
    const int*   objid         = (const int*)  d_in[3];
    const float* feats         = (const float*)d_in[4];

    const int B      = in_sizes[3];
    const int M      = in_sizes[0] / 2;
    const int npoint = in_sizes[2] / B;
    const int MAXP   = 128;
    const int C      = (out_size - (3 * B + 1)) / (B * (MAXP + 1));
    const int PTOT   = in_sizes[4] / C;
    const int kwords32 = npoint / 32;
    const int kwords64 = npoint / 64;

    unsigned long long pcW  = (unsigned long long)PTOT * kwords32;
    unsigned long long refW = (unsigned long long)B * kwords32;
    unsigned* pcbits  = (unsigned*)d_ws;
    unsigned* refbits = pcbits + pcW;
    int* interArr = (int*)(refbits + refW);
    int* unionArr = interArr + PTOT;
    unsigned long long wsWords = pcW + refW + 2ull * PTOT;

    float* out = (float*)d_out;
    const int selFeatOff = B * MAXP * C;
    const int selIdxOff  = selFeatOff + B * C;
    const int offOff     = selIdxOff + B;
    const int goodOff    = offOff + (B + 1);

    sg_init_kernel<<<2048, 256, 0, stream>>>(pcbits, wsWords, out, out_size);

    int totalW = (B * npoint) >> 5;
    sg_refbits_kernel<<<(totalW + 255) / 256, 256, 0, stream>>>(labels, objid, refbits,
                                                                totalW, npoint);

    sg_scatter_kernel<<<(M + 255) / 256, 256, 0, stream>>>(proposals_idx, pcbits,
                                                           M, npoint, kwords32);

    const int kchunk = 32;
    dim3 g3((PTOT + 15) / 16, (kwords64 + kchunk - 1) / kchunk);
    sg_iou_wmma_kernel<<<g3, 32, 0, stream>>>(pcbits, refbits, pes,
                                              interArr, unionArr,
                                              PTOT, B, kwords64, kchunk);

    sg_clus_copy_kernel<<<(PTOT * C + 255) / 256, 256, 0, stream>>>(feats, pes, out,
                                                                    B, PTOT, C, MAXP);

    sg_finalize_kernel<<<B, 128, 0, stream>>>(refbits, interArr, unionArr, feats, pes, out,
                                              B, C, npoint,
                                              selFeatOff, selIdxOff, offOff, goodOff);
}
